// LatentActionIDM_13048110645380
// MI455X (gfx1250) — compile-verified, matrix operand order checked by reading, not run on hardware
//
#include <hip/hip_runtime.h>
#include <hip/hip_bf16.h>

// ---------------- problem constants ----------------
#define NROWS 32768   // B*T
#define DD    256     // state dim
#define EE    128     // embedding dim
#define KCB   1024    // codebook size
#define HH    512     // MLP hidden

#ifndef USE_TDM
#define USE_TDM 1
#endif
#if USE_TDM && __has_builtin(__builtin_amdgcn_tensor_load_to_lds)
#define HAVE_TDM 1
#else
#define HAVE_TDM 0
#endif

typedef __attribute__((ext_vector_type(16))) __bf16          v16bf;
typedef __attribute__((ext_vector_type(16))) unsigned short  us16;
typedef __attribute__((ext_vector_type(8)))  unsigned short  us8;
typedef __attribute__((ext_vector_type(8)))  float           v8f;
typedef __attribute__((ext_vector_type(4)))  float           v4f;
typedef __attribute__((ext_vector_type(2)))  float           v2f;
typedef __attribute__((ext_vector_type(4)))  unsigned int    u32x4;
typedef __attribute__((ext_vector_type(8)))  int             i32x8;
typedef __attribute__((ext_vector_type(4)))  int             i32x4;

// ---------------- helpers ----------------
__device__ __forceinline__ unsigned short f2bf(float f) {
  unsigned int u = __builtin_bit_cast(unsigned int, f);
  u += 0x7FFFu + ((u >> 16) & 1u);           // round-to-nearest-even
  return (unsigned short)(u >> 16);
}

__device__ __forceinline__ float gelu_f(float x) {
  const float c = 0.7978845608028654f;       // sqrt(2/pi)
  float t = c * (x + 0.044715f * x * x * x);
  return 0.5f * x * (1.0f + tanhf(t));
}

#if HAVE_TDM
__device__ __forceinline__ unsigned int lds_off_of(const void* p) {
  return (unsigned int)(unsigned long long)
         (__attribute__((address_space(3))) const void*)p;
}

// 1-D TDM copy: n_u16 bf16 elements from global to LDS (D# per ISA 08 §8.3/8.4)
// This toolchain's builtin is the 6-arg form:
// (u32x4 g0, i32x8 g1, i32x4 g2, i32x4 g3, i32x8 pad, i32 cpol)
__device__ __forceinline__ void tdm_load_u16_1d(void* lds_dst, const void* gsrc,
                                                unsigned n_u16) {
  unsigned long long ga = (unsigned long long)gsrc;
  u32x4 g0;
  g0[0] = 1u;                                            // count=1 (valid D#)
  g0[1] = lds_off_of(lds_dst);                           // lds_addr
  g0[2] = (unsigned int)(ga & 0xFFFFFFFFu);              // global_addr[31:0]
  g0[3] = (unsigned int)((ga >> 32) & 0x1FFFFFFu)        // global_addr[56:32]
        | (2u << 30);                                    // type=2 ("image")
  i32x8 g1;
  g1[0] = (int)(1u << 16);                               // data_size=1 (2 bytes)
  g1[1] = (int)((n_u16 & 0xFFFFu) << 16);                // tensor_dim0[15:0]
  g1[2] = (int)((n_u16 >> 16) & 0xFFFFu);                // tensor_dim0[31:16]
  g1[3] = (int)((n_u16 & 0xFFFFu) << 16);                // tile_dim0
  g1[4] = 0;                                             // tile_dim1/2 = 0 (1-D)
  g1[5] = (int)n_u16;                                    // tensor_dim0_stride lo
  g1[6] = 0;
  g1[7] = 0;
  i32x4 z4 = {0, 0, 0, 0};
  i32x8 z8 = {0, 0, 0, 0, 0, 0, 0, 0};
  __builtin_amdgcn_tensor_load_to_lds(g0, g1, z4, z4, z8, 0);
}
#endif

// cooperative stage of one contiguous chunk of n_u16 bf16 elements into LDS
__device__ __forceinline__ void stage_chunk(unsigned short* sB,
                                            const unsigned short* gsrc,
                                            int n_u16, int wave) {
#if HAVE_TDM
  if (wave == 0) {
    tdm_load_u16_1d(sB, gsrc, (unsigned)n_u16);
    __builtin_amdgcn_s_wait_tensorcnt(0);
  }
#else
  const us8* src = (const us8*)gsrc;
  us8*       dst = (us8*)sB;
  int total = n_u16 / 8;
  for (int i = threadIdx.x; i < total; i += 256) dst[i] = src[i];
#endif
  __syncthreads();
}

// 16x32 bf16 WMMA fragment: lane owns row/col (idx&15); its 16 values are two
// 8-element K-contiguous runs: [k0+half*8, +7] and [k0+16+half*8, +7] -> 2xB128.
__device__ __forceinline__ v16bf load_frag(const unsigned short* __restrict__ base,
                                           int row, int ld, int k0, int half) {
  const unsigned short* p = base + (size_t)row * ld + k0 + half * 8;
  us8 lo = *(const us8*)p;
  us8 hi = *(const us8*)(p + 16);
  us16 t = __builtin_shufflevector(lo, hi, 0, 1, 2, 3, 4, 5, 6, 7,
                                           8, 9, 10, 11, 12, 13, 14, 15);
  return __builtin_bit_cast(v16bf, t);
}

// ---------------- small utility kernels ----------------
__global__ void cast_f32_bf16(const float* __restrict__ src,
                              unsigned short* __restrict__ dst, int n) {
  for (int i = blockIdx.x * blockDim.x + threadIdx.x; i < n;
       i += gridDim.x * blockDim.x)
    dst[i] = f2bf(src[i]);
}

// src is R x C (row-major), dst is C x R (row-major): dst[c*R+r] = src[r*C+c]
__global__ void transpose_cast_bf16(const float* __restrict__ src,
                                    unsigned short* __restrict__ dst,
                                    int R, int C) {
  int total = R * C;
  for (int i = blockIdx.x * blockDim.x + threadIdx.x; i < total;
       i += gridDim.x * blockDim.x) {
    int c = i / R, r = i - c * R;          // coalesced writes
    dst[i] = f2bf(src[(size_t)r * C + c]);
  }
}

__global__ void transpose_f32(const float* __restrict__ src,
                              float* __restrict__ dst, int R, int C) {
  int total = R * C;
  for (int i = blockIdx.x * blockDim.x + threadIdx.x; i < total;
       i += gridDim.x * blockDim.x) {
    int c = i / R, r = i - c * R;
    dst[i] = src[(size_t)r * C + c];
  }
}

__global__ void init_accum(unsigned int* counts, float* loss_sum) {
  int i = blockIdx.x * blockDim.x + threadIdx.x;
  if (i < KCB) counts[i] = 0u;
  if (i == 0) *loss_sum = 0.0f;
}

__global__ void cnorm_kernel(const float* __restrict__ cbTf, float* __restrict__ cn) {
  int n = blockIdx.x * blockDim.x + threadIdx.x;
  if (n < KCB) {
    float s = 0.f;
#pragma unroll 8
    for (int e = 0; e < EE; ++e) { float v = cbTf[(size_t)n * EE + e]; s += v * v; }
    cn[n] = s;
  }
}

// ---------------- bf16 WMMA GEMM with TDM/LDS-staged weights ----------------
// A: NROWS x KD (row-major bf16). Bt: ND x KD (row-major = W transposed).
// Block: 8 waves x 16-row stripes. Per 64-col chunk: DMA Bt chunk to LDS,
// all waves consume via ds_load_b128. A stripe held in registers.
template <int KD, int ND>
__global__ void gemm_bf16_kernel(const unsigned short* __restrict__ A,
                                 const unsigned short* __restrict__ Bt,
                                 const float* __restrict__ bias,
                                 unsigned short* __restrict__ out_bf,
                                 float* __restrict__ out_f32,
                                 int gelu_on, int out_ld, int col_off) {
  __shared__ unsigned short sB[64 * KD];
  int lane = threadIdx.x & 31, wave = threadIdx.x >> 5;
  int row0 = (blockIdx.x * 8 + wave) * 16;
  int half = lane >> 4, n = lane & 15;

  // whole A stripe in registers: KD/32 fragments of 8 VGPRs
  v16bf a[KD / 32];
#pragma unroll
  for (int kt = 0; kt < KD / 32; ++kt)
    a[kt] = load_frag(A, row0 + n, KD, kt * 32, half);

  for (int ct0 = 0; ct0 < ND / 16; ct0 += 4) {
    stage_chunk(sB, Bt + (size_t)(ct0 * 16) * KD, 64 * KD, wave);

    v8f acc[4];
#pragma unroll
    for (int c = 0; c < 4; ++c) {
      float bn = bias[(ct0 + c) * 16 + n];
#pragma unroll
      for (int r = 0; r < 8; ++r) acc[c][r] = bn;
    }
#pragma unroll
    for (int kt = 0; kt < KD / 32; ++kt) {
#pragma unroll
      for (int c = 0; c < 4; ++c) {
        v16bf b = load_frag(sB, c * 16 + n, KD, kt * 32, half);
        acc[c] = __builtin_amdgcn_wmma_f32_16x16x32_bf16(
            false, a[kt], false, b, (short)0, acc[c], false, false);
      }
    }
#pragma unroll
    for (int c = 0; c < 4; ++c) {
      int col = col_off + (ct0 + c) * 16 + n;
#pragma unroll
      for (int r = 0; r < 8; ++r) {
        float v = acc[c][r];
        if (gelu_on) v = gelu_f(v);
        size_t o = (size_t)(row0 + r + half * 8) * out_ld + col;
        if (out_f32) out_f32[o] = v;
        if (out_bf)  out_bf[o]  = f2bf(v);
      }
    }
    __syncthreads();  // protect sB before next chunk's DMA
  }
}

// ---------------- fused VQ: WMMA scores + argmin + one-hot + gather + loss ----
__global__ void vq_kernel(const float* __restrict__ latent,
                          const unsigned short* __restrict__ lat_bf,
                          const unsigned short* __restrict__ cbT,
                          const float* __restrict__ cbTf,
                          const float* __restrict__ cnorm,
                          float* __restrict__ zq, float* __restrict__ enc,
                          float* __restrict__ idx_out,
                          unsigned int* __restrict__ counts,
                          float* __restrict__ loss_sum) {
  __shared__ unsigned short sC[64 * EE];   // 16 KB codebook chunk
  int lane = threadIdx.x & 31, wave = threadIdx.x >> 5;
  int row0 = (blockIdx.x * 8 + wave) * 16;
  int half = lane >> 4, n = lane & 15;

  v16bf a[4];
#pragma unroll
  for (int kt = 0; kt < 4; ++kt)
    a[kt] = load_frag(lat_bf, row0 + n, EE, kt * 32, half);

  float bestd[8]; int besti[8];
#pragma unroll
  for (int r = 0; r < 8; ++r) { bestd[r] = 3.4e38f; besti[r] = 0; }

  // score = cnorm[k] - 2*f.c  (||f||^2 per-row constant -> argmin-invariant)
  for (int cc = 0; cc < KCB / 64; ++cc) {
    stage_chunk(sC, cbT + (size_t)(cc * 64) * EE, 64 * EE, wave);
#pragma unroll
    for (int t = 0; t < 4; ++t) {
      int col0 = cc * 64 + t * 16;
      v8f acc = {0.f, 0.f, 0.f, 0.f, 0.f, 0.f, 0.f, 0.f};
#pragma unroll
      for (int kt = 0; kt < 4; ++kt) {
        v16bf b = load_frag(sC, t * 16 + n, EE, kt * 32, half);
        acc = __builtin_amdgcn_wmma_f32_16x16x32_bf16(
            false, a[kt], false, b, (short)0, acc, false, false);
      }
      int col = col0 + n;
      float cn = cnorm[col];
#pragma unroll
      for (int r = 0; r < 8; ++r) {
        float sc = cn - 2.0f * acc[r];
        if (sc < bestd[r] || (sc == bestd[r] && col < besti[r])) {
          bestd[r] = sc; besti[r] = col;
        }
      }
    }
    __syncthreads();
  }

  // butterfly argmin across the 16 lanes of each half (wave32)
#pragma unroll
  for (int r = 0; r < 8; ++r) {
#pragma unroll
    for (int off = 1; off < 16; off <<= 1) {
      float od = __shfl_xor(bestd[r], off, 32);
      int   oi = __shfl_xor(besti[r], off, 32);
      if (od < bestd[r] || (od == bestd[r] && oi < besti[r])) {
        bestd[r] = od; besti[r] = oi;
      }
    }
  }

  float lsum = 0.f;
#pragma unroll
  for (int m = 0; m < 16; ++m) {
    int ii = __shfl(besti[m & 7], (m & 8) ? 16 : 0, 32);  // row m argmin
    if (lane == 0) {
      idx_out[row0 + m] = (float)ii;
      atomicAdd(&counts[ii], 1u);
    }
    // one-hot row: streaming nontemporal b64 stores, coalesced
    float* erow = enc + (size_t)(row0 + m) * KCB;
#pragma unroll
    for (int j = 0; j < 16; ++j) {
      int base = j * 64 + lane * 2;
      v2f z;
      z[0] = (base     == ii) ? 1.0f : 0.0f;
      z[1] = (base + 1 == ii) ? 1.0f : 0.0f;
      __builtin_nontemporal_store(z, (v2f*)(erow + base));
    }
    // z_q = gathered code row (contiguous in cbTf); commit-loss partial
    v4f q  = *(const v4f*)(cbTf   + (size_t)ii * EE + lane * 4);
    v4f lt = *(const v4f*)(latent + (size_t)(row0 + m) * EE + lane * 4);
    __builtin_nontemporal_store(q, (v4f*)(zq + (size_t)(row0 + m) * EE + lane * 4));
#pragma unroll
    for (int c = 0; c < 4; ++c) { float dd = q[c] - lt[c]; lsum += dd * dd; }
  }
#pragma unroll
  for (int off = 16; off >= 1; off >>= 1) lsum += __shfl_xor(lsum, off, 32);
  if (lane == 0) atomicAdd(loss_sum, lsum);
}

__global__ void finalize_kernel(const unsigned int* __restrict__ counts,
                                const float* __restrict__ loss_sum,
                                float* loss_out, float* perp_out) {
  __shared__ float sh[256];
  float s = 0.f;
  for (int k = threadIdx.x; k < KCB; k += 256) {
    float p = (float)counts[k] * (1.0f / (float)NROWS);
    s += p * logf(p + 1e-10f);
  }
  sh[threadIdx.x] = s; __syncthreads();
  for (int st = 128; st > 0; st >>= 1) {
    if (threadIdx.x < st) sh[threadIdx.x] += sh[threadIdx.x + st];
    __syncthreads();
  }
  if (threadIdx.x == 0) {
    *perp_out = expf(-sh[0]);
    *loss_out = 0.25f * (*loss_sum) / ((float)NROWS * (float)EE);
  }
}

// ---------------- host launch ----------------
extern "C" void kernel_launch(void* const* d_in, const int* in_sizes, int n_in,
                              void* d_out, int out_size, void* d_ws, size_t ws_size,
                              hipStream_t stream) {
  const float* states      = (const float*)d_in[0];
  const float* next_states = (const float*)d_in[1];
  const float* W_s = (const float*)d_in[2];  const float* b_s = (const float*)d_in[3];
  const float* W1  = (const float*)d_in[4];  const float* b1  = (const float*)d_in[5];
  const float* W2  = (const float*)d_in[6];  const float* b2  = (const float*)d_in[7];
  const float* W3  = (const float*)d_in[8];  const float* b3  = (const float*)d_in[9];
  const float* codebook = (const float*)d_in[10];

  char* ws = (char*)d_ws;
  const size_t off_stbf  = 0;                                    // 16 MB
  const size_t off_nsbf  = off_stbf + (size_t)NROWS * DD * 2;    // 16 MB
  const size_t off_h1    = 0;                                    // 32 MB (alias)
  const size_t off_x     = off_nsbf + (size_t)NROWS * DD * 2;    // 16 MB
  const size_t off_h2    = off_x + (size_t)NROWS * (2 * EE) * 2; // 32 MB
  const size_t off_lat   = off_h2 + (size_t)NROWS * HH * 2;      // 16 MB (f32)
  const size_t off_latbf = off_lat + (size_t)NROWS * EE * 4;     // 8 MB
  const size_t off_WsT   = off_latbf + (size_t)NROWS * EE * 2;
  const size_t off_W1T   = off_WsT + (size_t)DD * EE * 2;
  const size_t off_W2T   = off_W1T + (size_t)(2 * EE) * HH * 2;
  const size_t off_W3T   = off_W2T + (size_t)HH * HH * 2;
  const size_t off_cbT   = off_W3T + (size_t)HH * EE * 2;
  const size_t off_cbTf  = off_cbT + (size_t)KCB * EE * 2;
  const size_t off_cn    = off_cbTf + (size_t)KCB * EE * 4;
  const size_t off_cnt   = off_cn + (size_t)KCB * 4;
  const size_t off_ls    = off_cnt + (size_t)KCB * 4;

  unsigned short* stbf  = (unsigned short*)(ws + off_stbf);
  unsigned short* nsbf  = (unsigned short*)(ws + off_nsbf);
  unsigned short* xbf   = (unsigned short*)(ws + off_x);
  unsigned short* h1bf  = (unsigned short*)(ws + off_h1);
  unsigned short* h2bf  = (unsigned short*)(ws + off_h2);
  float*          lat   = (float*)(ws + off_lat);
  unsigned short* latbf = (unsigned short*)(ws + off_latbf);
  unsigned short* WsT   = (unsigned short*)(ws + off_WsT);
  unsigned short* W1T   = (unsigned short*)(ws + off_W1T);
  unsigned short* W2T   = (unsigned short*)(ws + off_W2T);
  unsigned short* W3T   = (unsigned short*)(ws + off_W3T);
  unsigned short* cbT   = (unsigned short*)(ws + off_cbT);
  float*          cbTf  = (float*)(ws + off_cbTf);
  float*          cn    = (float*)(ws + off_cn);
  unsigned int*   cnt   = (unsigned int*)(ws + off_cnt);
  float*          ls    = (float*)(ws + off_ls);

  float* out = (float*)d_out;
  float* zq     = out;                                     // N*EE
  float* loss_o = out + (size_t)NROWS * EE;                // 1
  float* perp_o = loss_o + 1;                              // 1
  float* enc    = perp_o + 1;                              // N*KCB
  float* idx_o  = enc + (size_t)NROWS * KCB;               // N

  // 1) activation casts + weight/codebook transposed casts (N-major -> b128 frags)
  cast_f32_bf16<<<2048, 256, 0, stream>>>(states,      stbf, NROWS * DD);
  cast_f32_bf16<<<2048, 256, 0, stream>>>(next_states, nsbf, NROWS * DD);
  transpose_cast_bf16<<<64,  256, 0, stream>>>(W_s, WsT, DD, EE);
  transpose_cast_bf16<<<256, 256, 0, stream>>>(W1,  W1T, 2 * EE, HH);
  transpose_cast_bf16<<<512, 256, 0, stream>>>(W2,  W2T, HH, HH);
  transpose_cast_bf16<<<128, 256, 0, stream>>>(W3,  W3T, HH, EE);
  transpose_cast_bf16<<<256, 256, 0, stream>>>(codebook, cbT,  EE, KCB);
  transpose_f32<<<256, 256, 0, stream>>>(codebook, cbTf, EE, KCB);

  init_accum<<<4, 256, 0, stream>>>(cnt, ls);
  cnorm_kernel<<<4, 256, 0, stream>>>(cbTf, cn);

  const int GB = NROWS / (16 * 8);  // 256 blocks, 8 waves, 16 rows/wave
  // 2) encoders -> gelu(concat) written into x
  gemm_bf16_kernel<DD, EE><<<GB, 256, 0, stream>>>(stbf, WsT, b_s, xbf, nullptr,
                                                   1, 2 * EE, 0);
  gemm_bf16_kernel<DD, EE><<<GB, 256, 0, stream>>>(nsbf, WsT, b_s, xbf, nullptr,
                                                   1, 2 * EE, EE);
  // 3) MLP
  gemm_bf16_kernel<2 * EE, HH><<<GB, 256, 0, stream>>>(xbf,  W1T, b1, h1bf, nullptr,
                                                       1, HH, 0);
  gemm_bf16_kernel<HH, HH><<<GB, 256, 0, stream>>>(h1bf, W2T, b2, h2bf, nullptr,
                                                   1, HH, 0);
  gemm_bf16_kernel<HH, EE><<<GB, 256, 0, stream>>>(h2bf, W3T, b3, latbf, lat,
                                                   0, EE, 0);
  // 4) VQ: scores via WMMA, argmin, one-hot, gather, loss partials
  vq_kernel<<<GB, 256, 0, stream>>>(lat, latbf, cbT, cbTf, cn,
                                    zq, enc, idx_o, cnt, ls);
  // 5) scalars
  finalize_kernel<<<1, 256, 0, stream>>>(cnt, ls, loss_o, perp_o);
}
